// DisplacementTensors_57707180589706
// MI455X (gfx1250) — matrix operand order repaired and checked by reading, not compiled, not run on hardware
//
#include <hip/hip_runtime.h>
#include <math.h>

// ---------------------------------------------------------------------------
// DisplacementTensors on MI455X (gfx1250, wave32, WMMA)
//
//   k0: convert w1,w2,w3,wv (f32) -> f16 into d_ws
//   k1: zero d_out
//   k2: per 16-edge tile (one wave32 each):
//         enc(8) + Linear(8->96) in f32 VALU -> LDS f16 tile
//         h1 = leaky(h0 @ w1^T + b1)   : 18x v_wmma_f32_16x16x32_f16
//         h2 = leaky(h1 @ w2^T + b2)   : 18x WMMA
//         rad = h2 @ w3^T              : 18x WMMA -> global_atomic_add A_a[src]
//         radv = rad @ wv^T            : 12x WMMA -> global_atomic_add out_v[src]*rv
// out_v = wv @ segsum(rad x rv) == segsum((wv @ rad) x rv)  (linearity).
// All global accesses forced to addrspace(1) (global_load/global_atomic, not
// flat), all tile accesses forced to addrspace(3) (ds_load_b128/ds_store_b16).
// ---------------------------------------------------------------------------

#define DIM_A 96
#define DIM_V 64
#define WAVES 8

typedef __attribute__((ext_vector_type(16))) _Float16 v16h;
typedef __attribute__((ext_vector_type(8)))  _Float16 v8h;
typedef __attribute__((ext_vector_type(8)))  float    v8f;
typedef __attribute__((ext_vector_type(4)))  float    v4f;

// address-space qualified aliases (generic -> AS cast is legal for pointers
// that really live in that space; forces global_* / ds_* instead of flat_*)
typedef __attribute__((address_space(1))) const _Float16 gch;
typedef __attribute__((address_space(1))) const v8h      gcv8h;
typedef __attribute__((address_space(1))) const float    gcf;
typedef __attribute__((address_space(1))) const v4f      gcv4f;
typedef __attribute__((address_space(1))) const int      gci;
typedef __attribute__((address_space(1))) float          gf;
typedef __attribute__((address_space(3))) const _Float16 lch;
typedef __attribute__((address_space(3))) const v8h      lcv8h;
typedef __attribute__((address_space(3))) _Float16       lh;

__device__ __forceinline__ float ldg_f(const float* p) { return *(gcf*)p; }
__device__ __forceinline__ v4f   ldg_f4(const float* p) { return *(gcv4f*)p; }
__device__ __forceinline__ int   ldg_i(const int* p)   { return *(gci*)p; }
__device__ __forceinline__ void  atomic_add_g(float* p, float v) {
    (void)__hip_atomic_fetch_add((gf*)p, v, __ATOMIC_RELAXED, __HIP_MEMORY_SCOPE_AGENT);
}

__device__ __forceinline__ v16h pack16_g(gch* p0, gch* p1) {
    v8h a = *(gcv8h*)p0;           // global_load_b128
    v8h b = *(gcv8h*)p1;
    v16h r;
#pragma unroll
    for (int i = 0; i < 8; ++i) { r[i] = a[i]; r[i + 8] = b[i]; }
    return r;
}

__device__ __forceinline__ v16h pack16_l(lch* p0, lch* p1) {
    v8h a = *(lcv8h*)p0;           // ds_load_b128
    v8h b = *(lcv8h*)p1;
    v16h r;
#pragma unroll
    for (int i = 0; i < 8; ++i) { r[i] = a[i]; r[i + 8] = b[i]; }
    return r;
}

__device__ __forceinline__ v8f wmma_f16(v16h a, v16h b, v8f c) {
    return __builtin_amdgcn_wmma_f32_16x16x32_f16(
        false, a, false, b, (short)0, c, false, false);
}

// Load the 3 K-step A-fragments (16 edges x 96 feats f16 tile in LDS).
// A layout (16x32 f16): lane row M = lane%16; 16 consecutive K halves at
// 32*ks + 16*(lane>=16).
__device__ __forceinline__ void load_a3(lch* t, int lol, int hi, v16h a[3]) {
    lch* base = t + lol * DIM_A + hi * 16;
#pragma unroll
    for (int ks = 0; ks < 3; ++ks)
        a[ks] = pack16_l(base + ks * 32, base + ks * 32 + 8);
}

// One hidden layer: dst = leaky(src @ W^T + bias); W f16 row-major [96][96].
// B layout (32x16 f16): lane col N = n0+lane%16, 16 consecutive K halves at
// 32*ks + 16*(lane>=16).
__device__ __forceinline__ void mlp_layer(lch* hsrc, lh* hdst, gch* W,
                                          const float* bias, int lol, int hi) {
    v16h a[3];
    load_a3(hsrc, lol, hi, a);
#pragma unroll
    for (int n0 = 0; n0 < DIM_A; n0 += 16) {
        const float bv = ldg_f(bias + n0 + lol);        // C[M][N] = bias[N]
        v8f c;
#pragma unroll
        for (int r = 0; r < 8; ++r) c[r] = bv;
        gch* wr = W + (n0 + lol) * DIM_A + hi * 16;
#pragma unroll
        for (int ks = 0; ks < 3; ++ks)
            c = wmma_f16(a[ks], pack16_g(wr + ks * 32, wr + ks * 32 + 8), c);
#pragma unroll
        for (int r = 0; r < 8; ++r) {                    // LeakyReLU + f16 restore
            float v = c[r];
            v = (v >= 0.0f) ? v : 0.1f * v;
            hdst[(r + hi * 8) * DIM_A + n0 + lol] = (_Float16)v;
        }
    }
}

// --- prep kernels -----------------------------------------------------------
__global__ void dt_convert_weights(const float* __restrict__ w1,
                                   const float* __restrict__ w2,
                                   const float* __restrict__ w3,
                                   const float* __restrict__ wv,
                                   _Float16* __restrict__ out) {
    const int i = blockIdx.x * blockDim.x + threadIdx.x;
    const int S = DIM_A * DIM_A;
    if (i < S)          out[i] = (_Float16)ldg_f(w1 + i);
    else if (i < 2 * S) out[i] = (_Float16)ldg_f(w2 + i - S);
    else if (i < 3 * S) out[i] = (_Float16)ldg_f(w3 + i - 2 * S);
    else if (i < 3 * S + DIM_V * DIM_A)
                        out[i] = (_Float16)ldg_f(wv + i - 3 * S);
}

__global__ void dt_zero(float* __restrict__ p, int n) {
    const int i = blockIdx.x * blockDim.x + threadIdx.x;
    if (i < n) *(gf*)(p + i) = 0.0f;
}

// --- main fused edge kernel --------------------------------------------------
__global__ __launch_bounds__(32 * WAVES)
void dt_edge_kernel(const float* __restrict__ r_ij,
                    const float* __restrict__ w0, const float* __restrict__ b0,
                    const float* __restrict__ b1, const float* __restrict__ b2,
                    const _Float16* __restrict__ wh1,
                    const _Float16* __restrict__ wh2,
                    const _Float16* __restrict__ wh3,
                    const _Float16* __restrict__ whv,
                    const int* __restrict__ srcIdx,
                    float* __restrict__ Aa,      // [N, 96]
                    float* __restrict__ outV,    // [N, 64, 3]
                    int E) {
    __shared__ __attribute__((aligned(16))) _Float16 tileS[WAVES][2][16 * DIM_A];
    __shared__ float rvs[WAVES][16][3];
    __shared__ int   srcs[WAVES][16];

    const int lane = threadIdx.x & 31;
    const int w    = threadIdx.x >> 5;
    const int lol  = lane & 15;
    const int hi   = lane >> 4;
    const int ebase = (blockIdx.x * WAVES + w) * 16;

    // warm L2 for the next block's edge tile (gfx1250 global_prefetch path)
    __builtin_prefetch(r_ij + (size_t)(ebase + 16 * WAVES) * 3 + lane, 0, 3);

    lh* tile0 = (lh*)&tileS[w][0][0];
    lh* tile1 = (lh*)&tileS[w][1][0];

    // ---------------- layer 0: encode + Linear(8->96), f32 VALU --------------
    {
        const int e  = lol;
        const int eg = ebase + e;
        float x0 = 0.f, x1 = 0.f, x2 = 0.f;
        if (eg < E) {
            x0 = ldg_f(r_ij + eg * 3 + 0);
            x1 = ldg_f(r_ij + eg * 3 + 1);
            x2 = ldg_f(r_ij + eg * 3 + 2);
        }
        const float rn = sqrtf(x0 * x0 + x1 * x1 + x2 * x2);
        const float u  = rn * 4.0f;                     // 8 / R0, R0 = 2
        float enc[8];
#pragma unroll
        for (int k = 0; k < 8; ++k) enc[k] = fmaxf(0.0f, 1.0f - fabsf(u - (float)k));
        if (hi == 0) {
            const float nx = 3.5f * rn;                 // |r_ij * 7/R0|
            const float s  = tanhf(nx) / fmaxf(nx, 1e-12f);
            rvs[w][e][0] = 3.5f * x0 * s;
            rvs[w][e][1] = 3.5f * x1 * s;
            rvs[w][e][2] = 3.5f * x2 * s;
            srcs[w][e]   = (eg < E) ? ldg_i(srcIdx + eg) : 0;
        }
        const int fbase = hi * 48;
        for (int f = fbase; f < fbase + 48; ++f) {
            const v4f wa = ldg_f4(w0 + f * 8);
            const v4f wb = ldg_f4(w0 + f * 8 + 4);
            float acc = ldg_f(b0 + f);
#pragma unroll
            for (int k = 0; k < 4; ++k) acc += enc[k] * wa[k];
#pragma unroll
            for (int k = 0; k < 4; ++k) acc += enc[k + 4] * wb[k];
            tile0[e * DIM_A + f] = (_Float16)acc;
        }
    }
    __syncthreads();

    // ---------------- layers 1 & 2 -------------------------------------------
    mlp_layer((lch*)tile0, tile1, (gch*)wh1, b1, lol, hi);
    __syncthreads();
    mlp_layer((lch*)tile1, tile0, (gch*)wh2, b2, lol, hi);
    __syncthreads();

    // ---------------- layer 3: rad = h @ w3^T, A_a scatter -------------------
    {
        v16h a[3];
        load_a3((lch*)tile0, lol, hi, a);
#pragma unroll
        for (int n0 = 0; n0 < DIM_A; n0 += 16) {
            v8f c;
#pragma unroll
            for (int r = 0; r < 8; ++r) c[r] = 0.0f;
            gch* wr = (gch*)wh3 + (n0 + lol) * DIM_A + hi * 16;
#pragma unroll
            for (int ks = 0; ks < 3; ++ks)
                c = wmma_f16(a[ks], pack16_g(wr + ks * 32, wr + ks * 32 + 8), c);
#pragma unroll
            for (int r = 0; r < 8; ++r) {
                const int el = r + hi * 8;
                const float v = c[r];
                tile1[el * DIM_A + n0 + lol] = (_Float16)v;  // rad for wv GEMM
                if (ebase + el < E)
                    atomic_add_g(&Aa[srcs[w][el] * DIM_A + n0 + lol], v);
            }
        }
    }
    __syncthreads();

    // ---------------- radv = rad @ wv^T, out_v scatter -----------------------
    {
        v16h a[3];
        load_a3((lch*)tile1, lol, hi, a);
#pragma unroll
        for (int n0 = 0; n0 < DIM_V; n0 += 16) {
            v8f c;
#pragma unroll
            for (int r = 0; r < 8; ++r) c[r] = 0.0f;
            gch* wr = (gch*)whv + (n0 + lol) * DIM_A + hi * 16;
#pragma unroll
            for (int ks = 0; ks < 3; ++ks)
                c = wmma_f16(a[ks], pack16_g(wr + ks * 32, wr + ks * 32 + 8), c);
#pragma unroll
            for (int r = 0; r < 8; ++r) {
                const int el = r + hi * 8;
                if (ebase + el < E) {
                    const int n = srcs[w][el];
                    const float v = c[r];
                    float* p = &outV[(n * DIM_V + n0 + lol) * 3];
                    atomic_add_g(p + 0, v * rvs[w][el][0]);
                    atomic_add_g(p + 1, v * rvs[w][el][1]);
                    atomic_add_g(p + 2, v * rvs[w][el][2]);
                }
            }
        }
    }
}

// ---------------------------------------------------------------------------
extern "C" void kernel_launch(void* const* d_in, const int* in_sizes, int n_in,
                              void* d_out, int out_size, void* d_ws, size_t ws_size,
                              hipStream_t stream) {
    const float* r_ij = (const float*)d_in[0];
    const float* w0   = (const float*)d_in[1];
    const float* b0   = (const float*)d_in[2];
    const float* w1   = (const float*)d_in[3];
    const float* b1   = (const float*)d_in[4];
    const float* w2   = (const float*)d_in[5];
    const float* b2   = (const float*)d_in[6];
    const float* w3   = (const float*)d_in[7];
    const float* wv   = (const float*)d_in[8];
    const int*   src  = (const int*)d_in[9];

    const int E = in_sizes[0] / 3;
    const int N = out_size / (DIM_A + DIM_V * 3);        // 288 floats per node

    float* out  = (float*)d_out;
    float* Aa   = out;                                    // [N, 96]
    float* outV = out + (size_t)N * DIM_A;                // [N, 64, 3]

    _Float16* wh  = (_Float16*)d_ws;
    _Float16* wh1 = wh;
    _Float16* wh2 = wh + DIM_A * DIM_A;
    _Float16* wh3 = wh + 2 * DIM_A * DIM_A;
    _Float16* whv = wh + 3 * DIM_A * DIM_A;

    const int convN = 3 * DIM_A * DIM_A + DIM_V * DIM_A;  // 33792
    dt_convert_weights<<<(convN + 255) / 256, 256, 0, stream>>>(w1, w2, w3, wv, wh);
    dt_zero<<<(out_size + 255) / 256, 256, 0, stream>>>(out, out_size);

    const int tiles  = (E + 15) / 16;
    const int blocks = (tiles + WAVES - 1) / WAVES;
    dt_edge_kernel<<<blocks, 32 * WAVES, 0, stream>>>(
        r_ij, w0, b0, b1, b2, wh1, wh2, wh3, whv, src, Aa, outV, E);
}
